// SoftTreeVectorized_56942676410676
// MI455X (gfx1250) — compile-verified, matrix operand order checked by reading, not run on hardware
//
#include <hip/hip_runtime.h>

// ---------------------------------------------------------------------------
// Soft decision tree, fused single kernel for MI455X (gfx1250, wave32, WMMA)
//   g   = sigmoid(x @ W + b)          : (B,128)x(128,63)  -> WMMA bf16
//   p   = tree expansion over 6 levels: per-row scalar fp32 (in registers)
//   out = p @ leaves                  : (B,64)x(64,32)    -> WMMA bf16
// Everything for a 256-row tile stays in LDS; only x is read and out written
// to HBM (~168 MB total -> ~7.2us floor at 23.3 TB/s).
// ---------------------------------------------------------------------------

#define TILE_B   256
#define NTHREADS 256
#define D_DIM    128
#define N_NODES  64     // 63 real internal nodes, padded to 64
#define LEAF_D   32
#define XP       136    // bf16 row stride for xA / Wt   (68 dwords, bank-safe)
#define GP       67     // f32  row stride for g         (odd -> bank-safe)
#define PP       72     // bf16 row stride for pB / leavesT (36 dwords)

typedef __attribute__((ext_vector_type(16))) __bf16 v16bf;
typedef __attribute__((ext_vector_type(8)))  float  v8f;

union Frag16 {
  v16bf v;
  uint4 q[2];
};

struct __align__(16) Smem {
  union {                                    // phase A/B: xA ; phase C/D: g
    unsigned short xA[TILE_B * XP];          // 69632 B
    float          g[TILE_B * GP];           // 68608 B
  } r1;
  union {                                    // phase B: Wt ; phase D/E: pB
    unsigned short Wt[N_NODES * XP];         // 17408 B
    unsigned short pB[TILE_B * PP];          // 36864 B
  } r2;
  unsigned short leavesT[LEAF_D * PP];       //  4608 B
  float bias[N_NODES];                       //   256 B
};                                           // ~109 KB total (<= 320 KB/WGP)

// ---- f32 -> bf16 helpers: prefer HW packed convert, fall back to RNE bits --
__device__ __forceinline__ unsigned short f2bf(float f) {
  unsigned int u = __float_as_uint(f);
  u += 0x7FFFu + ((u >> 16) & 1u);           // round-to-nearest-even
  return (unsigned short)(u >> 16);
}

#if __has_builtin(__builtin_amdgcn_cvt_pk_bf16_f32)
typedef __attribute__((ext_vector_type(2))) __bf16 v2bf;
__device__ __forceinline__ unsigned int pack_bf16(float a, float b) {
  v2bf r = __builtin_amdgcn_cvt_pk_bf16_f32(a, b);   // v_cvt_pk_bf16_f32
  return __builtin_bit_cast(unsigned int, r);
}
#else
__device__ __forceinline__ unsigned int pack_bf16(float a, float b) {
  return (unsigned int)f2bf(a) | ((unsigned int)f2bf(b) << 16);
}
#endif

// ---- fast sigmoid: prefer v_tanh_f32 (TRANS op on CDNA5), else exp+rcp ----
__device__ __forceinline__ float fast_sigmoid(float v) {
#if __has_builtin(__builtin_amdgcn_tanhf)
  return fmaf(0.5f, __builtin_amdgcn_tanhf(0.5f * v), 0.5f);
#elif __has_builtin(__builtin_amdgcn_rcpf)
  return __builtin_amdgcn_rcpf(1.0f + __expf(-v));
#else
  return 1.0f / (1.0f + __expf(-v));
#endif
}

__global__ __launch_bounds__(NTHREADS)
void soft_tree_fused(const float* __restrict__ x,
                     const float* __restrict__ W,
                     const float* __restrict__ b,
                     const float* __restrict__ leaves,
                     float* __restrict__ out)
{
  __shared__ Smem sm;

  const int tid     = threadIdx.x;
  const int lane    = tid & 31;
  const int wave    = tid >> 5;
  const int mrow    = lane & 15;             // fragment row/col within tile
  const int koff    = (lane >> 4) << 3;      // 0 for lanes 0-15, 8 for 16-31
  const int rowBase = blockIdx.x * TILE_B;

  // ---------------- Phase A: stage inputs into LDS (bf16) -----------------
  // x tile: 256 rows x 128 cols, coalesced float4 loads
  for (int i = tid; i < TILE_B * (D_DIM / 4); i += NTHREADS) {
    int r  = i >> 5;                         // 32 float4 per row
    int c4 = i & 31;
    float4 v = ((const float4*)x)[(size_t)(rowBase + r) * (D_DIM / 4) + c4];
    *(uint2*)&sm.r1.xA[r * XP + c4 * 4] =
        make_uint2(pack_bf16(v.x, v.y), pack_bf16(v.z, v.w));
  }
  // W^T: Wt[n][k] = W[k*63 + n], padded col 63 = 0; packed pairs along k
  for (int i = tid; i < N_NODES * (D_DIM / 2); i += NTHREADS) {
    int n = i >> 6, k = (i & 63) * 2;
    float w0 = (n < 63) ? W[k * 63 + n]       : 0.0f;
    float w1 = (n < 63) ? W[(k + 1) * 63 + n] : 0.0f;
    *(unsigned int*)&sm.r2.Wt[n * XP + k] = pack_bf16(w0, w1);
  }
  // leaves^T: leavesT[n][k] = leaves[k*32 + n]; packed pairs along k
  for (int i = tid; i < LEAF_D * 32; i += NTHREADS) {
    int n = i >> 5, k = (i & 31) * 2;
    *(unsigned int*)&sm.leavesT[n * PP + k] =
        pack_bf16(leaves[k * LEAF_D + n], leaves[(k + 1) * LEAF_D + n]);
  }
  if (tid < N_NODES) sm.bias[tid] = (tid < 63) ? b[tid] : 0.0f;
  __syncthreads();

  // ---------------- Phase B: GEMM1 via WMMA bf16 (g-pre-activation) -------
  // Each wave owns 32 rows = 2 strips of 16. 4 N-tiles x 4 K-steps each.
  Frag16 afrag[2][4];
  #pragma unroll
  for (int s = 0; s < 2; ++s) {
    int r0 = wave * 32 + s * 16;
    #pragma unroll
    for (int kk = 0; kk < 4; ++kk) {
      int base = (r0 + mrow) * XP + kk * 32 + koff;
      afrag[s][kk].q[0] = *(const uint4*)&sm.r1.xA[base];
      afrag[s][kk].q[1] = *(const uint4*)&sm.r1.xA[base + 16];
    }
  }
  v8f acc[2][4];
  #pragma unroll
  for (int nt = 0; nt < 4; ++nt) {
    Frag16 bfrag[4];
    #pragma unroll
    for (int kk = 0; kk < 4; ++kk) {
      int base = (nt * 16 + mrow) * XP + kk * 32 + koff;
      bfrag[kk].q[0] = *(const uint4*)&sm.r2.Wt[base];
      bfrag[kk].q[1] = *(const uint4*)&sm.r2.Wt[base + 16];
    }
    #pragma unroll
    for (int s = 0; s < 2; ++s) {
      v8f c = {0.f, 0.f, 0.f, 0.f, 0.f, 0.f, 0.f, 0.f};
      #pragma unroll
      for (int kk = 0; kk < 4; ++kk) {
        c = __builtin_amdgcn_wmma_f32_16x16x32_bf16(
                false, afrag[s][kk].v, false, bfrag[kk].v,
                (short)0, c, false, false);
      }
      acc[s][nt] = c;
    }
  }
  __syncthreads();   // all xA / Wt reads done; r1 may now be rewritten as g

  // ---------------- Phase C: bias + sigmoid, scatter g to LDS -------------
  #pragma unroll
  for (int s = 0; s < 2; ++s) {
    int rb = wave * 32 + s * 16 + koff;      // koff doubles as +8 row shift
    #pragma unroll
    for (int nt = 0; nt < 4; ++nt) {
      int   n  = nt * 16 + mrow;
      float bv = sm.bias[n];
      #pragma unroll
      for (int i = 0; i < 8; ++i) {
        sm.r1.g[(rb + i) * GP + n] = fast_sigmoid(acc[s][nt][i] + bv);
      }
    }
  }
  __syncthreads();

  // ---------------- Phase D: per-row tree expansion (fp32, in regs) -------
  {
    const float* gr = &sm.r1.g[tid * GP];
    float p[64];
    p[0] = 1.0f;
    int off = 0;
    #pragma unroll
    for (int l = 0; l < 6; ++l) {
      int cnt = 1 << l;
      #pragma unroll
      for (int j = cnt - 1; j >= 0; --j) {   // descending: safe in-place
        float pj = p[j];
        float gv = gr[off + j];
        float a  = gv * pj;
        p[2 * j]     = a;
        p[2 * j + 1] = pj - a;               // (1-g)*p
      }
      off += cnt;
    }
    unsigned short* pr = &sm.r2.pB[tid * PP];
    #pragma unroll
    for (int j = 0; j < 64; j += 2) {
      *(unsigned int*)&pr[j] = pack_bf16(p[j], p[j + 1]);
    }
  }
  __syncthreads();

  // ---------------- Phase E: GEMM2 via WMMA bf16 (p @ leaves) -------------
  Frag16 lfrag[2][2];
  #pragma unroll
  for (int nt = 0; nt < 2; ++nt)
    #pragma unroll
    for (int kk = 0; kk < 2; ++kk) {
      int base = (nt * 16 + mrow) * PP + kk * 32 + koff;
      lfrag[nt][kk].q[0] = *(const uint4*)&sm.leavesT[base];
      lfrag[nt][kk].q[1] = *(const uint4*)&sm.leavesT[base + 16];
    }
  float* outp = out + (size_t)rowBase * LEAF_D;
  #pragma unroll
  for (int s = 0; s < 2; ++s) {
    int r0 = wave * 32 + s * 16;
    Frag16 pf[2];
    #pragma unroll
    for (int kk = 0; kk < 2; ++kk) {
      int base = (r0 + mrow) * PP + kk * 32 + koff;
      pf[kk].q[0] = *(const uint4*)&sm.r2.pB[base];
      pf[kk].q[1] = *(const uint4*)&sm.r2.pB[base + 16];
    }
    #pragma unroll
    for (int nt = 0; nt < 2; ++nt) {
      v8f c = {0.f, 0.f, 0.f, 0.f, 0.f, 0.f, 0.f, 0.f};
      #pragma unroll
      for (int kk = 0; kk < 2; ++kk) {
        c = __builtin_amdgcn_wmma_f32_16x16x32_bf16(
                false, pf[kk].v, false, lfrag[nt][kk].v,
                (short)0, c, false, false);
      }
      int col = nt * 16 + mrow;
      int rb  = r0 + koff;
      #pragma unroll
      for (int i = 0; i < 8; ++i)
        outp[(size_t)(rb + i) * LEAF_D + col] = c[i];
    }
  }
}

extern "C" void kernel_launch(void* const* d_in, const int* in_sizes, int n_in,
                              void* d_out, int out_size, void* d_ws, size_t ws_size,
                              hipStream_t stream) {
  const float* x      = (const float*)d_in[0];
  const float* W      = (const float*)d_in[1];
  const float* b      = (const float*)d_in[2];
  const float* leaves = (const float*)d_in[3];
  float* out          = (float*)d_out;

  const int B = in_sizes[0] / D_DIM;         // 262144
  dim3 grid(B / TILE_B);                     // 1024 blocks
  dim3 block(NTHREADS);                      // 8 waves (wave32)
  hipLaunchKernelGGL(soft_tree_fused, grid, block, 0, stream,
                     x, W, b, leaves, out);
}
